// MSSA_80161269612831
// MI455X (gfx1250) — matrix operand order repaired and checked by reading, not compiled, not run on hardware
//
#include <hip/hip_runtime.h>
#include <math.h>

// ---------------------------------------------------------------------------
// Types for CDNA5 WMMA (wave32): A/B = 16 halves (8 VGPRs), C/D = 8 floats.
// ---------------------------------------------------------------------------
typedef __attribute__((ext_vector_type(16))) _Float16 v16h;
typedef __attribute__((ext_vector_type(8)))  float    v8f;
typedef __attribute__((ext_vector_type(4)))  float    v4f;
typedef __attribute__((ext_vector_type(2)))  unsigned int v2u;

union Frag {
    v16h v;
    _Float16 h[16];
    unsigned int u[8];
};

__device__ __forceinline__ v8f v8f_zero() {
    v8f z;
#pragma unroll
    for (int i = 0; i < 8; ++i) z[i] = 0.f;
    return z;
}

__device__ __forceinline__ v8f wmma_f16(v16h a, v16h b, v8f c) {
    return __builtin_amdgcn_wmma_f32_16x16x32_f16(
        /*neg_a=*/false, a, /*neg_b=*/false, b,
        /*c_mod=*/(short)0, c, /*reuse_a=*/false, /*reuse_b=*/false);
}

// --- gfx1250 async copy: global memory -> LDS, tracked by ASYNCcnt. --------
// LDS destination address = low 32 bits of the generic pointer (the LDS
// aperture maps addr[31:0] to the wave-relative LDS offset, ISA 10.2).
__device__ __forceinline__ unsigned lds_off(const void* p) {
    return (unsigned)(size_t)p;
}
__device__ __forceinline__ void async_b128(unsigned ldsaddr, const void* gptr) {
    asm volatile("global_load_async_to_lds_b128 %0, %1, off"
                 :: "v"(ldsaddr), "v"((unsigned long long)(size_t)gptr)
                 : "memory");
}
__device__ __forceinline__ void wait_async0() {
    asm volatile("s_wait_asynccnt 0x0" ::: "memory");
}

// A fragment (16x32, element (M,K) = base[M*ld + K]).
// lane = M + 16*(K>=16); half index = K & 15.
__device__ __forceinline__ v16h frag_a(const _Float16* base, int ld) {
    const int lane = threadIdx.x & 31;
    const int m  = lane & 15;
    const int kg = (lane >> 4) & 1;
    const _Float16* row = base + m * ld + 16 * kg;
    Frag f;
#pragma unroll
    for (int p = 0; p < 8; ++p) {
        const int k = 2 * p;
        f.u[p] = *(const unsigned int*)(row + k);
    }
    return f.v;
}

// B fragment (32x16, element (K,N) = base[N*ld + K]).
// lane = N + 16*(K>=16); half index = K & 15.
__device__ __forceinline__ v16h frag_b(const _Float16* base, int ld) {
    const int lane = threadIdx.x & 31;
    const int n  = lane & 15;
    const int kg = (lane >> 4) & 1;
    const _Float16* row = base + n * ld + 16 * kg;
    Frag f;
#pragma unroll
    for (int p = 0; p < 8; ++p)
        f.u[p] = *(const unsigned int*)(row + 2 * p);
    return f.v;
}

// B fragment, K-major (element (K,N) = base[K*ld + N]); used for P@V.
__device__ __forceinline__ v16h frag_b_strided(const _Float16* base, int ld) {
    const int lane = threadIdx.x & 31;
    const int n  = lane & 15;
    const int kg = (lane >> 4) & 1;
    Frag f;
#pragma unroll
    for (int i = 0; i < 16; ++i)
        f.h[i] = base[(i + 16 * kg) * ld + n];
    return f.v;
}

// ---------------------------------------------------------------------------
// f32 -> f16 conversion (one-time; makes every later tile stage a pure
// async DMA with no inner-loop VALU conversion).
// ---------------------------------------------------------------------------
__global__ void __launch_bounds__(256)
cvt16(const float* __restrict__ in, _Float16* __restrict__ out, int n4) {
    const int i = blockIdx.x * 256 + threadIdx.x;
    if (i < n4) {
        v4f v = *(const v4f*)(in + (size_t)i * 4);
        union { v2u u; _Float16 h[4]; } c;
        c.h[0] = (_Float16)v.x; c.h[1] = (_Float16)v.y;
        c.h[2] = (_Float16)v.z; c.h[3] = (_Float16)v.w;
        *(v2u*)(out + (size_t)i * 4) = c.u;
    }
}

// ---------------------------------------------------------------------------
// WMMA GEMM:  C[M,Nf] = A[M,K](f16) * W[Nf,K](f16)^T (+bias)(+res)(GELU)
// Block tile 128x64, 8 waves in 4(M)x2(N), each wave = 2x2 WMMA tiles.
// Double-buffered LDS fed by async global->LDS b128 copies.
// ---------------------------------------------------------------------------
#define GEMM_BM 128
#define GEMM_BN 64
#define GEMM_BK 32
#define LDSA 40   // padded LDS row stride in halves: 80 B (16B-aligned rows)

template <bool BIAS, bool RES, bool GELU, bool O16>
__global__ void __launch_bounds__(256)
gemm_wmma(const _Float16* __restrict__ A, const _Float16* __restrict__ W,
          const float* __restrict__ bias, const float* __restrict__ res,
          _Float16* __restrict__ o16, float* __restrict__ o32,
          int M, int Nf, int K) {
    __shared__ _Float16 As[2][GEMM_BM * LDSA];   // 2 x 10 KB
    __shared__ _Float16 Bs[2][GEMM_BN * LDSA];   // 2 x  5 KB

    const int tid  = threadIdx.x;
    const int wave = tid >> 5;
    const int wm = wave & 3;
    const int wn = wave >> 2;
    const int bm = blockIdx.y * GEMM_BM;
    const int bn = blockIdx.x * GEMM_BN;

    // Fixed per-thread copy slots (16B chunks = 8 halves).
    // A tile: 128 rows x 4 chunks = 512 slots -> 2 per thread.
    // B tile:  64 rows x 4 chunks = 256 slots -> 1 per thread.
    const int ar0 = tid >> 2, ac0 = (tid & 3) * 8;          // slot 0
    const int ar1 = (tid + 256) >> 2, ac1 = ac0;            // slot 1
    const int br  = tid >> 2, bc = (tid & 3) * 8;

    auto issue = [&](int k0, int buf) {
        async_b128(lds_off(&As[buf][ar0 * LDSA + ac0]),
                   A + (size_t)(bm + ar0) * K + k0 + ac0);
        async_b128(lds_off(&As[buf][ar1 * LDSA + ac1]),
                   A + (size_t)(bm + ar1) * K + k0 + ac1);
        async_b128(lds_off(&Bs[buf][br * LDSA + bc]),
                   W + (size_t)(bn + br) * K + k0 + bc);
    };

    v8f acc[2][2];
#pragma unroll
    for (int i = 0; i < 2; ++i)
#pragma unroll
        for (int j = 0; j < 2; ++j) acc[i][j] = v8f_zero();

    issue(0, 0);
    const int T = K / GEMM_BK;
    for (int kt = 0; kt < T; ++kt) {
        const int buf = kt & 1;
        wait_async0();            // my copies for tile kt are done
        __syncthreads();          // everyone's copies done, prev compute done
        if (kt + 1 < T) issue((kt + 1) * GEMM_BK, buf ^ 1);

        const v16h a0 = frag_a(&As[buf][(wm * 32 +  0) * LDSA], LDSA);
        const v16h a1 = frag_a(&As[buf][(wm * 32 + 16) * LDSA], LDSA);
        const v16h b0 = frag_b(&Bs[buf][(wn * 32 +  0) * LDSA], LDSA);
        const v16h b1 = frag_b(&Bs[buf][(wn * 32 + 16) * LDSA], LDSA);
        acc[0][0] = wmma_f16(a0, b0, acc[0][0]);
        acc[0][1] = wmma_f16(a0, b1, acc[0][1]);
        acc[1][0] = wmma_f16(a1, b0, acc[1][0]);
        acc[1][1] = wmma_f16(a1, b1, acc[1][1]);
        __syncthreads();          // done reading buf before it is re-filled
    }

    // Epilogue. C layout: lane = N + 16*(M>=8); vgpr r -> M = r + 8*(lane>>4).
    const int lane = tid & 31;
    const int cn = lane & 15;
    const int r0 = (lane >> 4) * 8;
#pragma unroll
    for (int i = 0; i < 2; ++i)
#pragma unroll
        for (int j = 0; j < 2; ++j) {
            const int gn = bn + wn * 32 + j * 16 + cn;
            const float bv = BIAS ? bias[gn] : 0.f;
#pragma unroll
            for (int r = 0; r < 8; ++r) {
                const int gm = bm + wm * 32 + i * 16 + r0 + r;
                float v = acc[i][j][r] + bv;
                if (RES)  v += res[(size_t)gm * Nf + gn];
                if (GELU) v = 0.5f * v * (1.f + erff(v * 0.70710678118654752f));
                if (O16)  o16[(size_t)gm * Nf + gn] = (_Float16)v;
                else      o32[(size_t)gm * Nf + gn] = v;
            }
        }
}

// ---------------------------------------------------------------------------
// Flash attention (q=k=v=w). Grid: (N/128, B*H). 8 waves x 16 query rows.
// 32-key strips: S = Q Kt (4 WMMA), online softmax, O += P V (4 WMMA).
// KV strip double-buffered in LDS via async copies, shared by all waves.
// ---------------------------------------------------------------------------
#define KVS 72    // 144 B rows (16B-aligned), conflict-free n*36 mod 64
#define PST 34

__global__ void __launch_bounds__(256)
attn_wmma(const _Float16* __restrict__ w, _Float16* __restrict__ o) {
    const int Nn = 2048;
    __shared__ _Float16 KV[2][32 * KVS];          // 2 x 4.5 KB
    __shared__ _Float16 Pst[8 * 16 * PST];        // 8.5 KB

    const int tid  = threadIdx.x;
    const int lane = tid & 31;
    const int wave = tid >> 5;
    const int b = blockIdx.y >> 3;
    const int h = blockIdx.y & 7;
    const int q0 = blockIdx.x * 128 + wave * 16;

    // Q fragments (dh 0-31 / 32-63) straight from global ([B,N,H*64], dh contig).
    const _Float16* qbase = w + ((size_t)(b * Nn + q0)) * 512 + h * 64;
    const v16h qa0 = frag_a(qbase,      512);
    const v16h qa1 = frag_a(qbase + 32, 512);

    // Per-thread KV copy slot: 32 rows x 8 chunks of 16B.
    const int kr = tid >> 3;        // 0..31
    const int ks = (tid & 7) * 8;   // halves 0..56
    const _Float16* kvg = w + ((size_t)(b * Nn + kr)) * 512 + h * 64 + ks;

    auto issue = [&](int kb, int buf) {
        async_b128(lds_off(&KV[buf][kr * KVS + ks]), kvg + (size_t)kb * 512);
    };

    v8f ov[4];
#pragma unroll
    for (int t = 0; t < 4; ++t) ov[t] = v8f_zero();
    float mrow[8], lrow[8];
#pragma unroll
    for (int r = 0; r < 8; ++r) { mrow[r] = -1e30f; lrow[r] = 0.f; }

    _Float16* myP = &Pst[wave * 16 * PST];
    const float sc = 0.125f;   // 64^-0.5

    issue(0, 0);
    for (int kb = 0; kb < Nn; kb += 32) {
        const int buf = (kb >> 5) & 1;
        wait_async0();
        __syncthreads();
        if (kb + 32 < Nn) issue(kb + 32, buf ^ 1);
        const _Float16* kv = KV[buf];

        // S strip: two 16x16 tiles over dh=64 (2 WMMA each).
        v8f s[2];
#pragma unroll
        for (int t = 0; t < 2; ++t) {
            v8f z = v8f_zero();
            z = wmma_f16(qa0, frag_b(&kv[(16 * t) * KVS +  0], KVS), z);
            z = wmma_f16(qa1, frag_b(&kv[(16 * t) * KVS + 32], KVS), z);
            s[t] = z;
        }

        // Online softmax over the strip's 32 keys.
        float alpha[8];
#pragma unroll
        for (int r = 0; r < 8; ++r) {
            const float t0 = s[0][r] * sc;
            const float t1 = s[1][r] * sc;
            float tm = fmaxf(t0, t1);
            tm = fmaxf(tm, __shfl_xor(tm, 1));
            tm = fmaxf(tm, __shfl_xor(tm, 2));
            tm = fmaxf(tm, __shfl_xor(tm, 4));
            tm = fmaxf(tm, __shfl_xor(tm, 8));
            const float mn = fmaxf(mrow[r], tm);
            alpha[r] = __expf(mrow[r] - mn);
            mrow[r] = mn;
            const float p0 = __expf(t0 - mn);
            const float p1 = __expf(t1 - mn);
            float rs = p0 + p1;
            rs += __shfl_xor(rs, 1);
            rs += __shfl_xor(rs, 2);
            rs += __shfl_xor(rs, 4);
            rs += __shfl_xor(rs, 8);
            lrow[r] = lrow[r] * alpha[r] + rs;
            const int row = (lane >> 4) * 8 + r;
            myP[row * PST +      (lane & 15)] = (_Float16)p0;
            myP[row * PST + 16 + (lane & 15)] = (_Float16)p1;
        }
#pragma unroll
        for (int t = 0; t < 4; ++t)
#pragma unroll
            for (int r = 0; r < 8; ++r) ov[t][r] *= alpha[r];

        // Intra-wave cross-lane RAW on the P stage.
        asm volatile("s_wait_dscnt 0" ::: "memory");

        const v16h pa = frag_a(myP, PST);
#pragma unroll
        for (int t = 0; t < 4; ++t)
            ov[t] = wmma_f16(pa, frag_b_strided(&kv[16 * t], KVS), ov[t]);

        __syncthreads();   // done reading kv buffer before re-fill
    }

    // Normalize, write O to [B,N,H*64] f16.
#pragma unroll
    for (int r = 0; r < 8; ++r) lrow[r] = 1.f / lrow[r];
#pragma unroll
    for (int t = 0; t < 4; ++t)
#pragma unroll
        for (int r = 0; r < 8; ++r) {
            const int gq = q0 + (lane >> 4) * 8 + r;
            o[((size_t)(b * Nn + gq)) * 512 + h * 64 + t * 16 + (lane & 15)] =
                (_Float16)(ov[t][r] * lrow[r]);
        }
}

// ---------------------------------------------------------------------------
// Launcher.
// ---------------------------------------------------------------------------
extern "C" void kernel_launch(void* const* d_in, const int* in_sizes, int n_in,
                              void* d_out, int out_size, void* d_ws, size_t ws_size,
                              hipStream_t stream) {
    (void)in_sizes; (void)n_in; (void)out_size; (void)ws_size;

    const float* x     = (const float*)d_in[0];
    const float* qkv_w = (const float*)d_in[1];
    const float* out_w = (const float*)d_in[2];
    const float* out_b = (const float*)d_in[3];
    const float* ff1_w = (const float*)d_in[4];
    const float* ff1_b = (const float*)d_in[5];
    const float* ff2_w = (const float*)d_in[6];
    const float* ff2_b = (const float*)d_in[7];

    const int M = 4 * 2048;
    const int D = 512;

    // Workspace (26 MB) with liveness-based aliasing:
    //   R0: x16 (step 1)            then h16 (steps 4-5)
    //   R1: w16 (steps 1-2)         then r16 (steps 3-4)
    //   R2: a16 (steps 2-3)
    char* ws = (char*)d_ws;
    const size_t seg = (size_t)M * D * sizeof(_Float16);   // 8 MB
    const size_t wsz = (size_t)D * D * sizeof(_Float16);   // 0.5 MB
    _Float16* x16  = (_Float16*)(ws);
    _Float16* h16  = (_Float16*)(ws);
    _Float16* w16  = (_Float16*)(ws + seg);
    _Float16* r16  = (_Float16*)(ws + seg);
    _Float16* a16  = (_Float16*)(ws + 2 * seg);
    _Float16* wq16 = (_Float16*)(ws + 3 * seg);
    _Float16* wo16 = (_Float16*)(ws + 3 * seg + wsz);
    _Float16* wf1  = (_Float16*)(ws + 3 * seg + 2 * wsz);
    _Float16* wf2  = (_Float16*)(ws + 3 * seg + 3 * wsz);

    // One-time f32 -> f16 packs (activations + weights).
    cvt16<<<dim3((M * D) / 1024), dim3(256), 0, stream>>>(x,     x16,  (M * D) / 4);
    cvt16<<<dim3((D * D) / 1024), dim3(256), 0, stream>>>(qkv_w, wq16, (D * D) / 4);
    cvt16<<<dim3((D * D) / 1024), dim3(256), 0, stream>>>(out_w, wo16, (D * D) / 4);
    cvt16<<<dim3((D * D) / 1024), dim3(256), 0, stream>>>(ff1_w, wf1,  (D * D) / 4);
    cvt16<<<dim3((D * D) / 1024), dim3(256), 0, stream>>>(ff2_w, wf2,  (D * D) / 4);

    const dim3 gg(D / GEMM_BN, M / GEMM_BM);   // (8, 64)
    const dim3 bb(256);

    // 1) w = x @ qkv_w^T -> f16
    gemm_wmma<false, false, false, true><<<gg, bb, 0, stream>>>(
        x16, wq16, nullptr, nullptr, w16, nullptr, M, D, D);

    // 2) flash attention (q=k=v=w) -> f16
    attn_wmma<<<dim3(2048 / 128, 4 * 8), bb, 0, stream>>>(w16, a16);

    // 3) out = attn @ out_w^T + out_b + x -> f16
    gemm_wmma<true, true, false, true><<<gg, bb, 0, stream>>>(
        a16, wo16, out_b, x, r16, nullptr, M, D, D);

    // 4) h = gelu(out @ ff1_w^T + ff1_b) -> f16
    gemm_wmma<true, false, true, true><<<gg, bb, 0, stream>>>(
        r16, wf1, ff1_b, nullptr, h16, nullptr, M, D, D);

    // 5) final = h @ ff2_w^T + ff2_b -> f32 d_out
    gemm_wmma<true, false, false, false><<<gg, bb, 0, stream>>>(
        h16, wf2, ff2_b, nullptr, nullptr, (float*)d_out, M, D, D);
}
